// LSTMSupertaggingModel_3642132267370
// MI455X (gfx1250) — matrix operand order repaired and checked by reading, not compiled, not run on hardware
//
#include <hip/hip_runtime.h>
#include <hip/hip_bf16.h>
#include <math.h>

// Problem sizes (fixed by the reference)
constexpr int Bsz = 64, Ssz = 512, Dsz = 768, Wsz = 256, Hsz = 384, Csz = 425;
constexpr int Gsz = 4 * Hsz;   // 1536 gate width
constexpr int H2  = 2 * Hsz;   // 768 concat width
constexpr int CP  = 432;       // C padded to multiple of 16

typedef __attribute__((ext_vector_type(16))) _Float16 v16h;
typedef __attribute__((ext_vector_type(8)))  _Float16 v8h;
typedef __attribute__((ext_vector_type(8)))  float    v8f;

// ---- WMMA fragment helpers (CDNA5 16x16x32 f16, wave32) -------------------
// A (16x32, MxK) layout: lanes 0-15 -> row=lane, K in {0..7,16..23};
// lanes 16-31 -> row=lane-16, K in {8..15,24..31}. Two 16B loads per lane.
__device__ __forceinline__ v16h load_a_frag(const _Float16* __restrict__ tile, int ld) {
  int lane = threadIdx.x & 31;
  int row  = lane & 15;
  int kb   = (lane >> 4) << 3;                 // 0 or 8
  const v8h* p0 = reinterpret_cast<const v8h*>(tile + (size_t)row * ld + kb);
  const v8h* p1 = reinterpret_cast<const v8h*>(tile + (size_t)row * ld + kb + 16);
  v8h lo = *p0, hi = *p1;
  return __builtin_shufflevector(lo, hi, 0,1,2,3,4,5,6,7,8,9,10,11,12,13,14,15);
}

// B fragments are pre-packed so each lane's 16 halfs are contiguous (32B load).
__device__ __forceinline__ v16h load_b_frag(const _Float16* __restrict__ pack, size_t tile) {
  int lane = threadIdx.x & 31;
  return *reinterpret_cast<const v16h*>(pack + tile * 512 + (size_t)lane * 16);
}

__device__ __forceinline__ v8f wmma16(v16h a, v16h b, v8f c) {
  return __builtin_amdgcn_wmma_f32_16x16x32_f16(false, a, false, b, (short)0, c, false, false);
}

__device__ __forceinline__ v8f zero8() {
  v8f z = {0.f,0.f,0.f,0.f,0.f,0.f,0.f,0.f};
  return z;
}

__device__ __forceinline__ float sigmoidf_(float x) { return 1.0f / (1.0f + __expf(-x)); }

// ---- 1) ragged segment-mean pool -> f16 activations -----------------------
__global__ void __launch_bounds__(256) pool_kernel(const float* __restrict__ f0,
                                                   const int* __restrict__ counts,
                                                   _Float16* __restrict__ xh) {
  __shared__ int scan[Wsz];
  __shared__ int scnt[Wsz];
  int b = blockIdx.x, t = threadIdx.x;
  int cnt = counts[b * Wsz + t];
  scnt[t] = cnt;
  scan[t] = cnt;
  __syncthreads();
  // Hillis-Steele inclusive scan over 256 entries
  for (int off = 1; off < Wsz; off <<= 1) {
    int add = (t >= off) ? scan[t - off] : 0;
    __syncthreads();
    scan[t] += add;
    __syncthreads();
  }
  const float*  fb = f0 + (size_t)b * Ssz * Dsz;
  _Float16*     xb = xh + (size_t)b * Ssz * Dsz;
  // pooled words [0, W): consecutive threads -> consecutive d (coalesced)
  for (int idx = t; idx < Wsz * Dsz; idx += 256) {
    int w = idx / Dsz, d = idx - w * Dsz;
    int st = scan[w] - scnt[w];
    float v = fb[(size_t)st * Dsz + d];
    if (scnt[w] == 2) v = 0.5f * (v + fb[(size_t)(st + 1) * Dsz + d]);
    xb[(size_t)w * Dsz + d] = (_Float16)v;
  }
  // tail [W, S): raw copy
  for (int idx = t; idx < (Ssz - Wsz) * Dsz; idx += 256) {
    int s = Wsz + idx / Dsz, d = idx % Dsz;
    xb[(size_t)s * Dsz + d] = (_Float16)fb[(size_t)s * Dsz + d];
  }
}

// ---- 2) pack W [N,K] row-major -> W^T as fragment-contiguous B tiles ------
__global__ void pack_b_kernel(const float* __restrict__ Wm, _Float16* __restrict__ out,
                              int N, int K, int Npad) {
  int KT = K >> 5, NT = Npad >> 4;
  int total = KT * NT * 512;
  for (int idx = blockIdx.x * blockDim.x + threadIdx.x; idx < total;
       idx += gridDim.x * blockDim.x) {
    int w  = idx & 511;
    int tl = idx >> 9;
    int kt = tl % KT, nt = tl / KT;            // tile index = nt*KT + kt
    int lane = w >> 4, i = w & 15;
    int k = (kt << 5) + ((i >> 3) << 4) + ((lane >> 4) << 3) + (i & 7);   // B(32x16): lanes16-31 -> K in {8..15,24..31}
    int n = (nt << 4) + (lane & 15);
    float v = (n < N) ? Wm[(size_t)n * K + k] : 0.f;
    out[idx] = (_Float16)v;
  }
}

__global__ void init_state_kernel(_Float16* __restrict__ hbuf, float* __restrict__ cbuf,
                                  unsigned* __restrict__ barrier_ctr) {
  int i = blockIdx.x * blockDim.x + threadIdx.x;
  if (i < 4 * Bsz * Hsz) hbuf[i] = (_Float16)0.f;   // 2 dirs x 2 ping-pong
  if (i < 2 * Bsz * Hsz) cbuf[i] = 0.f;             // 2 dirs
  if (i < 2) barrier_ctr[i] = 0u;                   // per-direction arrival counters
}

// ---- 3) input projection: xp[dir] = xh @ Wih^T + bih + bhh (f16 out) -----
__global__ void __launch_bounds__(256) xproj_kernel(
    const _Float16* __restrict__ xh,
    const _Float16* __restrict__ wpkf, const _Float16* __restrict__ wpkb,
    const float* __restrict__ bihf, const float* __restrict__ bhhf,
    const float* __restrict__ bihb, const float* __restrict__ bhhb,
    _Float16* __restrict__ xp) {
  const int KT = Dsz / 32;                      // 24
  int dir = blockIdx.z;
  const _Float16* wpk = dir ? wpkb : wpkf;
  const float* bih = dir ? bihb : bihf;
  const float* bhh = dir ? bhhb : bhhf;
  _Float16* out = xp + (size_t)dir * Bsz * Ssz * Gsz;

  int wave = threadIdx.x >> 5, lane = threadIdx.x & 31;
  int m0 = blockIdx.x * 128 + (wave & 1) * 64;  // 64x64 tile per wave
  int n0 = blockIdx.y * 256 + (wave >> 1) * 64;

  v8f acc[4][4];
#pragma unroll
  for (int i = 0; i < 4; ++i)
#pragma unroll
    for (int j = 0; j < 4; ++j) acc[i][j] = zero8();

  for (int kt = 0; kt < KT; ++kt) {
    v16h af[4], bf[4];
#pragma unroll
    for (int i = 0; i < 4; ++i)
      af[i] = load_a_frag(xh + (size_t)(m0 + i * 16) * Dsz + kt * 32, Dsz);
#pragma unroll
    for (int j = 0; j < 4; ++j)
      bf[j] = load_b_frag(wpk, (size_t)(n0 / 16 + j) * KT + kt);
#pragma unroll
    for (int i = 0; i < 4; ++i)
#pragma unroll
      for (int j = 0; j < 4; ++j) acc[i][j] = wmma16(af[i], bf[j], acc[i][j]);
  }

  int ln = lane & 15, lh = lane >> 4;
#pragma unroll
  for (int j = 0; j < 4; ++j) {
    int ncol = n0 + j * 16 + ln;
    float bias = bih[ncol] + bhh[ncol];
#pragma unroll
    for (int i = 0; i < 4; ++i)
#pragma unroll
      for (int v = 0; v < 8; ++v) {
        int m = m0 + i * 16 + v + 8 * lh;       // C/D layout: M = v + 8*(lane>=16)
        out[(size_t)m * Gsz + ncol] = (_Float16)(acc[i][j][v] + bias);
      }
  }
}

// ---- 4) persistent bidirectional LSTM scan --------------------------------
// Grid = 48 resident blocks: blocks [0,24) = forward, [24,48) = backward.
// Each block owns 16 hidden units x 4 gates x 64 batch. Per step: WMMA GEMM
// gates = h @ Whh^T (into LDS, + xp which has biases folded), cell update,
// then an agent-scope atomic barrier over the 24 blocks of this direction.
__global__ void __launch_bounds__(256) lstm_scan_kernel(
    const _Float16* __restrict__ xp,      // [2][B*S, 4H] f16
    const _Float16* __restrict__ whh_pk,  // [2] packed B tiles
    _Float16* __restrict__ hbuf,          // [2 dirs][2 ping-pong][B*H] f16
    float* __restrict__ cbuf,             // [2 dirs][B*H] f32
    _Float16* __restrict__ A2,            // [B, S, 2H] f16 concat output
    unsigned* __restrict__ barrier_ctr) { // [2] arrival counters
  __shared__ float gl[64 * 64];                 // gates staging, 16 KB
  const int KT = Hsz / 32;                      // 12
  int dir = blockIdx.x / 24;
  int wgj = blockIdx.x % 24;                    // hidden-unit chunk

  const _Float16* xpD  = xp + (size_t)dir * Bsz * Ssz * Gsz;
  const _Float16* wpkD = whh_pk + (size_t)dir * Hsz * Gsz;
  _Float16* hbD = hbuf + (size_t)dir * 2 * Bsz * Hsz;
  float*    cst = cbuf + (size_t)dir * Bsz * Hsz;
  unsigned* ctr = barrier_ctr + dir;
  int dirOff = dir * Hsz;

  int wave = threadIdx.x >> 5, lane = threadIdx.x & 31;
  int gate = wave & 3;                          // i,f,g,o
  int mh   = wave >> 2;                         // batch half
  int nt   = gate * 24 + wgj;                   // n-tile in [0,96)
  int ln = lane & 15, lh = lane >> 4;
  int ncol = gate * Hsz + wgj * 16 + ln;

  for (int t = 0; t < Ssz; ++t) {
    int s = dir ? (Ssz - 1 - t) : t;
    const _Float16* h_in  = hbD + (size_t)(t & 1) * Bsz * Hsz;
    _Float16*       h_out = hbD + (size_t)((t + 1) & 1) * Bsz * Hsz;

    v8f acc0 = zero8(), acc1 = zero8();
#pragma unroll
    for (int kt = 0; kt < KT; ++kt) {
      v16h b  = load_b_frag(wpkD, (size_t)nt * KT + kt);
      v16h a0 = load_a_frag(h_in + (size_t)(mh * 32) * Hsz + kt * 32, Hsz);
      v16h a1 = load_a_frag(h_in + (size_t)(mh * 32 + 16) * Hsz + kt * 32, Hsz);
      acc0 = wmma16(a0, b, acc0);
      acc1 = wmma16(a1, b, acc1);
    }

#pragma unroll
    for (int v = 0; v < 8; ++v) {
      int b0 = mh * 32 + v + 8 * lh;
      int b1 = b0 + 16;
      gl[b0 * 64 + gate * 16 + ln] = acc0[v] + (float)xpD[((size_t)b0 * Ssz + s) * Gsz + ncol];
      gl[b1 * 64 + gate * 16 + ln] = acc1[v] + (float)xpD[((size_t)b1 * Ssz + s) * Gsz + ncol];
    }
    __syncthreads();

    // cell update: 64 batch x 16 hidden units
    for (int it = threadIdx.x; it < 64 * 16; it += 256) {
      int bb = it >> 4, jj = it & 15;
      float ig = gl[bb * 64 +       jj];
      float fg = gl[bb * 64 + 16 +  jj];
      float gg = gl[bb * 64 + 32 +  jj];
      float og = gl[bb * 64 + 48 +  jj];
      int hidx = bb * Hsz + wgj * 16 + jj;
      float c = cst[hidx];
      c = sigmoidf_(fg) * c + sigmoidf_(ig) * tanhf(gg);
      float h = sigmoidf_(og) * tanhf(c);
      cst[hidx] = c;
      _Float16 hh = (_Float16)h;
      h_out[hidx] = hh;
      A2[((size_t)bb * Ssz + s) * H2 + dirOff + wgj * 16 + jj] = hh;
    }

    // inter-block barrier (this direction only): release h_out, arrive, wait
    __threadfence();                 // make this thread's h_out stores agent-visible
    __syncthreads();                 // all threads done + fenced before signal
    if (threadIdx.x == 0) {
      __hip_atomic_fetch_add(ctr, 1u, __ATOMIC_RELEASE, __HIP_MEMORY_SCOPE_AGENT);
      unsigned target = 24u * (unsigned)(t + 1);
      while (__hip_atomic_load(ctr, __ATOMIC_ACQUIRE, __HIP_MEMORY_SCOPE_AGENT) < target)
        __builtin_amdgcn_s_sleep(1);
    }
    __syncthreads();
    __threadfence();                 // invalidate stale lines before next h_in read
  }
}

// ---- 5) classifier: out = [hf|hb] @ Wlin^T + blin (f32 out) --------------
__global__ void __launch_bounds__(256) gemm_out_kernel(
    const _Float16* __restrict__ A2, const _Float16* __restrict__ wpk,
    const float* __restrict__ blin, float* __restrict__ out) {
  const int KT = H2 / 32;                       // 24
  int wave = threadIdx.x >> 5, lane = threadIdx.x & 31;
  int m0  = blockIdx.x * 256 + wave * 32;       // 32x48 per wave
  int nt0 = blockIdx.y * 3;

  v8f acc[2][3];
#pragma unroll
  for (int i = 0; i < 2; ++i)
#pragma unroll
    for (int j = 0; j < 3; ++j) acc[i][j] = zero8();

  for (int kt = 0; kt < KT; ++kt) {
    v16h a0 = load_a_frag(A2 + (size_t)m0 * H2 + kt * 32, H2);
    v16h a1 = load_a_frag(A2 + (size_t)(m0 + 16) * H2 + kt * 32, H2);
#pragma unroll
    for (int j = 0; j < 3; ++j) {
      v16h b = load_b_frag(wpk, (size_t)(nt0 + j) * KT + kt);
      acc[0][j] = wmma16(a0, b, acc[0][j]);
      acc[1][j] = wmma16(a1, b, acc[1][j]);
    }
  }

  int ln = lane & 15, lh = lane >> 4;
#pragma unroll
  for (int j = 0; j < 3; ++j) {
    int ncol = (nt0 + j) * 16 + ln;
    if (ncol < Csz) {
      float bias = blin[ncol];
#pragma unroll
      for (int i = 0; i < 2; ++i)
#pragma unroll
        for (int v = 0; v < 8; ++v) {
          int m = m0 + i * 16 + v + 8 * lh;
          out[(size_t)m * Csz + ncol] = acc[i][j][v] + bias;
        }
    }
  }
}

// ---------------------------------------------------------------------------
extern "C" void kernel_launch(void* const* d_in, const int* in_sizes, int n_in,
                              void* d_out, int out_size, void* d_ws, size_t ws_size,
                              hipStream_t stream) {
  const float* f0     = (const float*)d_in[0];
  const int*   counts = (const int*)  d_in[1];
  const float* Wih_f  = (const float*)d_in[2];
  const float* Whh_f  = (const float*)d_in[3];
  const float* bih_f  = (const float*)d_in[4];
  const float* bhh_f  = (const float*)d_in[5];
  const float* Wih_b  = (const float*)d_in[6];
  const float* Whh_b  = (const float*)d_in[7];
  const float* bih_b  = (const float*)d_in[8];
  const float* bhh_b  = (const float*)d_in[9];
  const float* Wlin   = (const float*)d_in[10];
  const float* blin   = (const float*)d_in[11];
  float* out = (float*)d_out;

  // Workspace carve-out (all offsets 256B-aligned)
  char* base = (char*)d_ws;
  size_t off = 0;
  auto take = [&](size_t bytes) -> char* {
    char* r = base + off;
    off = (off + bytes + 255) & ~(size_t)255;
    return r;
  };
  _Float16* xh      = (_Float16*)take((size_t)Bsz * Ssz * Dsz * 2);       // 48 MB
  _Float16* wih_pk  = (_Float16*)take((size_t)2 * Dsz * Gsz * 2);        // 4.5 MB
  _Float16* whh_pk  = (_Float16*)take((size_t)2 * Hsz * Gsz * 2);        // 2.25 MB
  _Float16* wlin_pk = (_Float16*)take((size_t)H2 * CP * 2);              // 0.66 MB
  _Float16* xp      = (_Float16*)take((size_t)2 * Bsz * Ssz * Gsz * 2);  // 192 MB
  _Float16* A2      = (_Float16*)take((size_t)Bsz * Ssz * H2 * 2);       // 48 MB
  _Float16* hbuf    = (_Float16*)take((size_t)4 * Bsz * Hsz * 2);        // dirs x pp
  float*    cbuf    = (float*)   take((size_t)2 * Bsz * Hsz * 4);
  unsigned* barctr  = (unsigned*)take(2 * sizeof(unsigned));
  (void)ws_size; (void)n_in; (void)in_sizes; (void)out_size;

  // 1) pool + f16 convert
  pool_kernel<<<Bsz, 256, 0, stream>>>(f0, counts, xh);

  // 2) pack weights into B-fragment layout; zero state + barrier counters
  pack_b_kernel<<<1024, 256, 0, stream>>>(Wih_f, wih_pk,                     Gsz, Dsz, Gsz);
  pack_b_kernel<<<1024, 256, 0, stream>>>(Wih_b, wih_pk + (size_t)Dsz * Gsz, Gsz, Dsz, Gsz);
  pack_b_kernel<<< 512, 256, 0, stream>>>(Whh_f, whh_pk,                     Gsz, Hsz, Gsz);
  pack_b_kernel<<< 512, 256, 0, stream>>>(Whh_b, whh_pk + (size_t)Hsz * Gsz, Gsz, Hsz, Gsz);
  pack_b_kernel<<< 256, 256, 0, stream>>>(Wlin,  wlin_pk,                    Csz, H2,  CP);
  init_state_kernel<<<(4 * Bsz * Hsz + 255) / 256, 256, 0, stream>>>(hbuf, cbuf, barctr);

  // 3) input projections (both directions), bias folded in
  xproj_kernel<<<dim3(Bsz * Ssz / 128, Gsz / 256, 2), 256, 0, stream>>>(
      xh, wih_pk, wih_pk + (size_t)Dsz * Gsz,
      bih_f, bhh_f, bih_b, bhh_b, xp);

  // 4) persistent bidirectional scan: ONE launch, 48 resident blocks,
  //    512 steps with per-direction atomic barriers between steps
  lstm_scan_kernel<<<48, 256, 0, stream>>>(xp, whh_pk, hbuf, cbuf, A2, barctr);

  // 5) classifier
  gemm_out_kernel<<<dim3(Bsz * Ssz / 256, CP / 48), 256, 0, stream>>>(
      A2, wlin_pk, blin, out);
}
